// EXGNN_26001732010523
// MI455X (gfx1250) — matrix operand order repaired and checked by reading, not compiled, not run on hardware
//
#include <hip/hip_runtime.h>
#include <hip/hip_bf16.h>
#include <math.h>

#define N0_   262144
#define N1_   65536
#define N2_   16384
#define NNET_ 200000
#define E0_   2097152
#define E1_   524288
#define E2_   131072
#define ECONN_ 800000

typedef __attribute__((ext_vector_type(16))) _Float16 v16h;
typedef __attribute__((ext_vector_type(8)))  float    v8f;

// ---------------------------------------------------------------------------
// WMMA f16 A-fragment loader: one 16x32 (MxK) chunk, f32 source, zero-padded.
// Lane L holds row L%16; element e -> k = kc*32 + laneHi*8 + e (+8 for e>=8).
// Group validity depends only on (kc, g) since K % 16 == 0 for all K used,
// so with K a template constant these guards fold away at compile time and
// the loads become straight-line global_load_b128 pairs.
// ---------------------------------------------------------------------------
template <int K>
__device__ inline v16h load_a16(const float* __restrict__ row, int kc, int laneHi) {
    v16h a;
#pragma unroll
    for (int g = 0; g < 2; ++g) {
        if (kc * 32 + g * 16 + 16 <= K) {   // compile-time when kc is unrolled
            int kk = kc * 32 + laneHi * 8 + g * 16;
            float4 v0 = *(const float4*)(row + kk);
            float4 v1 = *(const float4*)(row + kk + 4);
            int e = g * 8;
            a[e + 0] = (_Float16)v0.x; a[e + 1] = (_Float16)v0.y;
            a[e + 2] = (_Float16)v0.z; a[e + 3] = (_Float16)v0.w;
            a[e + 4] = (_Float16)v1.x; a[e + 5] = (_Float16)v1.y;
            a[e + 6] = (_Float16)v1.z; a[e + 7] = (_Float16)v1.w;
        } else {
            int e = g * 8;
#pragma unroll
            for (int q = 0; q < 8; ++q) a[e + q] = (_Float16)0.0f;
        }
    }
    return a;
}

// ---------------------------------------------------------------------------
// out = tanh(Xa @ Wa [+ Xb @ Wb] + bias), row-major, M % 16 == 0.
// One wave per 16-row x 16-col tile; grid.y = Dout/16; grid-stride over rows.
// K (compile-time) padded to KC*32 with zeros.  Emits v_wmma_f32_16x16x32_f16.
// ---------------------------------------------------------------------------
template <int K, bool DUAL, bool TANH>
__global__ __launch_bounds__(128) void sage_wmma_kernel(
    const float* __restrict__ Xa, const float* __restrict__ Xb,
    const float* __restrict__ Wa, const float* __restrict__ Wb,
    const float* __restrict__ bias, float* __restrict__ out,
    int M, int Dout) {
    constexpr int KC = (K + 31) / 32;
    const int wave   = threadIdx.x >> 5;
    const int lane   = threadIdx.x & 31;
    const int laneHi = lane >> 4;   // 0 or 1
    const int lcol   = lane & 15;
    const int col    = blockIdx.y * 16 + lcol;

    // Preload B fragments (weights), f32 -> f16, zero beyond K (setup only).
    // Lane n: col n%16, element e -> krow = kc*32 + (n/16)*8 + e (+8 for e>=8).
    v16h Ba[KC];
    v16h Bb[DUAL ? KC : 1];
#pragma unroll
    for (int kc = 0; kc < KC; ++kc) {
        v16h fa, fb;
#pragma unroll
        for (int e = 0; e < 16; ++e) {
            int krow = kc * 32 + laneHi * 8 + e + (e >= 8 ? 8 : 0);
            float wa = (krow < K) ? Wa[(size_t)krow * Dout + col] : 0.0f;
            fa[e] = (_Float16)wa;
            if (DUAL) {
                float wb = (krow < K) ? Wb[(size_t)krow * Dout + col] : 0.0f;
                fb[e] = (_Float16)wb;
            }
        }
        Ba[kc] = fa;
        if (DUAL) Bb[kc] = fb;
    }
    const float bv = bias[col];

    const int Mtiles  = M >> 4;
    const int tile0   = blockIdx.x * 4 + wave;
    const int tstride = gridDim.x * 4;
    for (int mt = tile0; mt < Mtiles; mt += tstride) {
        const int arow = mt * 16 + lcol;  // A row for this lane (both halves)
        const float* pa = Xa + (size_t)arow * K;
        v8f acc = {};
#pragma unroll
        for (int kc = 0; kc < KC; ++kc) {
            v16h a = load_a16<K>(pa, kc, laneHi);
            acc = __builtin_amdgcn_wmma_f32_16x16x32_f16(
                false, a, false, Ba[kc], (short)0, acc, false, false);
        }
        if (DUAL) {
            const float* pb = Xb + (size_t)arow * K;
#pragma unroll
            for (int kc = 0; kc < KC; ++kc) {
                v16h a = load_a16<K>(pb, kc, laneHi);
                acc = __builtin_amdgcn_wmma_f32_16x16x32_f16(
                    false, a, false, Bb[kc], (short)0, acc, false, false);
            }
        }
        // D layout: reg r, lane L -> row = r + (L/16)*8, col = L%16
#pragma unroll
        for (int r = 0; r < 8; ++r) {
            int orow = mt * 16 + laneHi * 8 + r;
            float v  = acc[r] + bv;
            if (TANH) v = tanhf(v);
            out[(size_t)orow * Dout + col] = v;
        }
    }
}

// ------------------------- irregular / elementwise -------------------------
__global__ void fill_f32(float* __restrict__ p, float v, long long n) {
    long long i = (long long)blockIdx.x * blockDim.x + threadIdx.x;
    if (i < n) p[i] = v;
}

__global__ void degree_accum(const int* __restrict__ dst, float* __restrict__ deg, int nE) {
    int e = blockIdx.x * blockDim.x + threadIdx.x;
    if (e < nE) atomicAdd(&deg[dst[e]], 1.0f);
}

__global__ void scatter_add_rows(const float* __restrict__ x, const int* __restrict__ src,
                                 const int* __restrict__ dst, float* __restrict__ agg,
                                 int nE, int D) {
    long long t = (long long)blockIdx.x * blockDim.x + threadIdx.x;
    if (t >= (long long)nE * D) return;
    int e = (int)(t / D);
    int j = (int)(t - (long long)e * D);
    int s = src ? src[e] : e;
    atomicAdd(&agg[(size_t)dst[e] * D + j], x[(size_t)s * D + j]);
}

__global__ void normalize_rows(float* __restrict__ agg, const float* __restrict__ deg,
                               int n, int D) {
    long long t = (long long)blockIdx.x * blockDim.x + threadIdx.x;
    if (t >= (long long)n * D) return;
    int i = (int)(t / D);
    agg[t] = agg[t] / fmaxf(deg[i], 1.0f);
}

__global__ void concat_gather(const float* __restrict__ up, const int* __restrict__ assign,
                              const float* __restrict__ low, float* __restrict__ out,
                              int n, int Dup, int Dlow) {
    int D = Dup + Dlow;
    long long t = (long long)blockIdx.x * blockDim.x + threadIdx.x;
    if (t >= (long long)n * D) return;
    int i = (int)(t / D);
    int j = (int)(t - (long long)i * D);
    float v;
    if (j < Dup) v = up[(size_t)assign[i] * Dup + j];
    else         v = low[(size_t)i * Dlow + (j - Dup)];
    out[t] = v;
}

__device__ inline void atomic_max_f32(float* addr, float val) {
    unsigned int* ua  = (unsigned int*)addr;
    unsigned int  cur = __float_as_uint(*addr);
    while (__uint_as_float(cur) < val) {
        unsigned int prev = atomicCAS(ua, cur, __float_as_uint(val));
        if (prev == cur) break;
        cur = prev;
    }
}

__global__ void seg_max_kernel(const float* __restrict__ x, int ldx, int coloff,
                               const int* __restrict__ src, const int* __restrict__ dst,
                               float* __restrict__ y, int nE, int D) {
    long long t = (long long)blockIdx.x * blockDim.x + threadIdx.x;
    if (t >= (long long)nE * D) return;
    int e = (int)(t / D);
    int j = (int)(t - (long long)e * D);
    float v = x[(size_t)src[e] * ldx + coloff + j];
    atomic_max_f32(&y[(size_t)dst[e] * D + j], v);
}

__global__ void build_xx(const float* __restrict__ ymax, const float* __restrict__ ymin,
                         const float* __restrict__ xnet, float* __restrict__ xx, int n) {
    long long t = (long long)blockIdx.x * blockDim.x + threadIdx.x;
    if (t >= (long long)n * 80) return;
    int i = (int)(t / 80);
    int j = (int)(t - (long long)i * 80);
    float v;
    if (j < 32) {
        v = ymax[(size_t)i * 32 + j];
        if (v < -2.0f) v = 0.0f;  // tanh outputs are in (-1,1); -inf => empty net
    } else if (j < 64) {
        v = ymin[(size_t)i * 32 + (j - 32)];
        if (v < -2.0f) v = 0.0f;
    } else {
        v = xnet[(size_t)i * 16 + (j - 64)];
    }
    xx[t] = v;
}

__global__ void mlp2_kernel(const float* __restrict__ h, const float* __restrict__ w2,
                            const float* __restrict__ b2, float* __restrict__ out, int n) {
    int i = blockIdx.x * blockDim.x + threadIdx.x;
    if (i >= n) return;
    const float4* hp = (const float4*)(h + (size_t)i * 64);
    float s = 0.f;
#pragma unroll
    for (int q = 0; q < 16; ++q) {
        float4 hv = hp[q];
        s += hv.x * w2[4 * q + 0] + hv.y * w2[4 * q + 1] +
             hv.z * w2[4 * q + 2] + hv.w * w2[4 * q + 3];
    }
    out[i] = s + b2[0];
}

// ------------------------------- launchers ---------------------------------
static inline unsigned nblk(long long n, int t) { return (unsigned)((n + t - 1) / t); }

static void launch_fill(float* p, float v, long long n, hipStream_t s) {
    fill_f32<<<nblk(n, 256), 256, 0, s>>>(p, v, n);
}

// scatter-mean: agg[dst] = mean over edges of x[src] (src==NULL -> src=e)
static void seg_mean(const float* x, const int* src, const int* dst, float* agg, float* deg,
                     int nE, int nOut, int D, hipStream_t s) {
    launch_fill(deg, 0.f, nOut, s);
    launch_fill(agg, 0.f, (long long)nOut * D, s);
    degree_accum<<<nblk(nE, 256), 256, 0, s>>>(dst, deg, nE);
    scatter_add_rows<<<nblk((long long)nE * D, 256), 256, 0, s>>>(x, src, dst, agg, nE, D);
    normalize_rows<<<nblk((long long)nOut * D, 256), 256, 0, s>>>(agg, deg, nOut, D);
}

template <int K, bool DUAL>
static void launch_sage(const float* Xa, const float* Xb, const float* Wa, const float* Wb,
                        const float* bias, float* out, int M, int Dout, hipStream_t s) {
    int Mtiles = M / 16;
    int gx = (Mtiles + 3) / 4;
    if (gx > 4096) gx = 4096;
    dim3 grid((unsigned)gx, (unsigned)(Dout / 16));
    sage_wmma_kernel<K, DUAL, true><<<grid, 128, 0, s>>>(Xa, Xb, Wa, Wb, bias, out, M, Dout);
}

extern "C" void kernel_launch(void* const* d_in, const int* in_sizes, int n_in,
                              void* d_out, int out_size, void* d_ws, size_t ws_size,
                              hipStream_t stream) {
    (void)in_sizes; (void)n_in; (void)out_size; (void)ws_size;
    const float* x0    = (const float*)d_in[0];
    const float* x_net = (const float*)d_in[1];
    const float* Ws0 = (const float*)d_in[2],  *Wn0 = (const float*)d_in[3],  *b0 = (const float*)d_in[4];
    const float* Ws1 = (const float*)d_in[5],  *Wn1 = (const float*)d_in[6],  *b1 = (const float*)d_in[7];
    const float* Ws2 = (const float*)d_in[8],  *Wn2 = (const float*)d_in[9],  *b2 = (const float*)d_in[10];
    const float* Ws3 = (const float*)d_in[11], *Wn3 = (const float*)d_in[12], *b3 = (const float*)d_in[13];
    const float* Ws4 = (const float*)d_in[14], *Wn4 = (const float*)d_in[15], *b4 = (const float*)d_in[16];
    const float* mlp_w1 = (const float*)d_in[17], *mlp_b1 = (const float*)d_in[18];
    const float* mlp_w2 = (const float*)d_in[19], *mlp_b2 = (const float*)d_in[20];
    const int* e0_src = (const int*)d_in[21], *e0_dst = (const int*)d_in[22];
    const int* e1_src = (const int*)d_in[23], *e1_dst = (const int*)d_in[24];
    const int* e2_src = (const int*)d_in[25], *e2_dst = (const int*)d_in[26];
    const int* assign01 = (const int*)d_in[27], *assign12 = (const int*)d_in[28];
    const int* conn_src = (const int*)d_in[29], *conn_dst = (const int*)d_in[30];
    float* out = (float*)d_out;
    float* ws  = (float*)d_ws;

    // Workspace layout (floats). Three big N0*64 regions reused phase-by-phase.
    float* R1  = ws;
    float* R2  = ws + (size_t)N0_ * 64;
    float* R3  = ws + (size_t)N0_ * 64 * 2;
    float* h1v = ws + (size_t)N0_ * 64 * 3;       // [N1,32] persists
    float* h2v = h1v + (size_t)N1_ * 32;          // [N2,32] persists
    float* h1b = h2v + (size_t)N2_ * 32;          // [N1,32] persists
    float* deg = h1b + (size_t)N1_ * 32;          // [N0] reused per level

    // ---- level 0: h0 = sage(x0, e0) ----
    float* agg0 = R1;                             // [N0,16]
    seg_mean(x0, e0_src, e0_dst, agg0, deg, E0_, N0_, 16, stream);
    float* h0 = R3;                               // [N0,32]
    launch_sage<16, true>(x0, agg0, Ws0, Wn0, b0, h0, N0_, 32, stream);

    // ---- pool 0->1, level 1 ----
    float* x1 = R2;                               // [N1,32]
    seg_mean(h0, nullptr, assign01, x1, deg, N0_, N1_, 32, stream);
    float* agg1 = R2 + (size_t)N1_ * 32;          // [N1,32]
    seg_mean(x1, e1_src, e1_dst, agg1, deg, E1_, N1_, 32, stream);
    launch_sage<32, true>(x1, agg1, Ws1, Wn1, b1, h1v, N1_, 32, stream);

    // ---- pool 1->2, level 2 ----  (x1/agg1 dead: reuse R2)
    float* x2 = R2;                               // [N2,32]
    seg_mean(h1v, nullptr, assign12, x2, deg, N1_, N2_, 32, stream);
    float* agg2 = R2 + (size_t)N2_ * 32;          // [N2,32]
    seg_mean(x2, e2_src, e2_dst, agg2, deg, E2_, N2_, 32, stream);
    launch_sage<32, true>(x2, agg2, Ws2, Wn2, b2, h2v, N2_, 32, stream);

    // ---- up 2->1: cat1 = [h2[assign12] | h1], sage with Ws3/Wn3 ----
    float* cat1 = R1;                             // [N1,64] (agg0 dead)
    concat_gather<<<nblk((long long)N1_ * 64, 256), 256, 0, stream>>>(
        h2v, assign12, h1v, cat1, N1_, 32, 32);
    float* aggc1 = R2;                            // [N1,64] (x2/agg2 dead)
    seg_mean(cat1, e1_src, e1_dst, aggc1, deg, E1_, N1_, 64, stream);
    launch_sage<64, true>(cat1, aggc1, Ws3, Wn3, b3, h1b, N1_, 32, stream);

    // ---- up 1->0: cat0 = [h1b[assign01] | h0], sage with Ws4/Wn4 ----
    float* cat0 = R1;                             // [N0,64] (cat1 dead)
    concat_gather<<<nblk((long long)N0_ * 64, 256), 256, 0, stream>>>(
        h1b, assign01, h0, cat0, N0_, 32, 32);
    float* aggc0 = R2;                            // [N0,64] (aggc1 dead)
    seg_mean(cat0, e0_src, e0_dst, aggc0, deg, E0_, N0_, 64, stream);
    float* h0b = R3;                              // [N0,64] (h0 dead after cat0)
    launch_sage<64, true>(cat0, aggc0, Ws4, Wn4, b4, h0b, N0_, 64, stream);

    // ---- per-net segment max over conn edges ----
    float* ymax = R1;                             // [NNET,32] (cat0 dead)
    float* ymin = R1 + (size_t)NNET_ * 32;        // [NNET,32]
    launch_fill(ymax, -INFINITY, (long long)NNET_ * 32, stream);
    launch_fill(ymin, -INFINITY, (long long)NNET_ * 32, stream);
    seg_max_kernel<<<nblk((long long)ECONN_ * 32, 256), 256, 0, stream>>>(
        h0b, 64, 0, conn_src, conn_dst, ymax, ECONN_, 32);
    seg_max_kernel<<<nblk((long long)ECONN_ * 32, 256), 256, 0, stream>>>(
        h0b, 64, 32, conn_src, conn_dst, ymin, ECONN_, 32);

    // ---- MLP head ----
    float* xx = R2;                               // [NNET,80] (aggc0 dead)
    build_xx<<<nblk((long long)NNET_ * 80, 256), 256, 0, stream>>>(ymax, ymin, x_net, xx, NNET_);
    float* hmlp = R3;                             // [NNET,64] (h0b dead)
    launch_sage<80, false>(xx, nullptr, mlp_w1, nullptr, mlp_b1, hmlp, NNET_, 64, stream);
    mlp2_kernel<<<nblk(NNET_, 256), 256, 0, stream>>>(hmlp, mlp_w2, mlp_b2, out, NNET_);
}
